// AttentionLayer_5119601016898
// MI455X (gfx1250) — compile-verified, matrix operand order checked by reading, not run on hardware
//
#include <hip/hip_runtime.h>
#include <hip/hip_bf16.h>
#include <stdint.h>

#define NROWS  100000
#define MEDGES 800000
#define HID    128
#define HEADS  8
#define DH     16
#define NTILES (NROWS / 16)   // 6250, exact

typedef __attribute__((ext_vector_type(16))) __bf16 v16bf;
typedef __attribute__((ext_vector_type(8)))  __bf16 v8bf;
typedef __attribute__((ext_vector_type(8)))  float  v8f;

// ---------------------------------------------------------------------------
// f32 -> bf16 conversion
// ---------------------------------------------------------------------------
__global__ __launch_bounds__(256) void f32_to_bf16_kernel(
    const float* __restrict__ src, __bf16* __restrict__ dst, int n) {
  int t = blockIdx.x * blockDim.x + threadIdx.x;
  if (t < n) dst[t] = (__bf16)src[t];
}

// ---------------------------------------------------------------------------
// GEMM: out[N,128] = X[N,128] @ W^T (+bias) (relu) (+resid), optional bf16 copy
// One wave computes a 16x128 output tile with 8 f32 accumulators and
// v_wmma_f32_16x16x32_bf16 (4 K-steps x 8 N-blocks = 32 WMMA issues).
// Fragment layout per ISA 7.12.2 (16-bit A 16x32): lanes 0-15 hold
// K {0..7,16..23}, lanes 16-31 hold K {8..15,24..31}; B mirrored with
// n = j*16 + (lane&15), reading W row-major (W[n][k]) => out = X @ W^T.
// ---------------------------------------------------------------------------
__global__ __launch_bounds__(128) void gemm_bf16_128(
    const __bf16* __restrict__ X, const __bf16* __restrict__ W,
    const float* __restrict__ bias, const float* __restrict__ resid,
    float* __restrict__ outf, __bf16* __restrict__ outh,
    int nTiles, int doRelu) {
  int wave = blockIdx.x * (blockDim.x >> 5) + (threadIdx.x >> 5);
  if (wave >= nTiles) return;
  int lane    = threadIdx.x & 31;
  int halfSel = lane >> 4;     // 0: lanes 0-15, 1: lanes 16-31
  int l16     = lane & 15;
  int row0    = wave * 16;

  v8f acc[8];
#pragma unroll
  for (int j = 0; j < 8; ++j) acc[j] = (v8f){0.f, 0.f, 0.f, 0.f, 0.f, 0.f, 0.f, 0.f};

  const __bf16* xrow = X + (size_t)(row0 + l16) * HID;

#pragma unroll
  for (int kb = 0; kb < 4; ++kb) {
    int k0 = kb * 32 + (halfSel ? 8 : 0);
    int k1 = k0 + 16;
    v8bf alo = *(const v8bf*)(xrow + k0);
    v8bf ahi = *(const v8bf*)(xrow + k1);
    v16bf a = __builtin_shufflevector(alo, ahi, 0, 1, 2, 3, 4, 5, 6, 7,
                                      8, 9, 10, 11, 12, 13, 14, 15);
#pragma unroll
    for (int j = 0; j < 8; ++j) {
      const __bf16* wrow = W + (size_t)(j * 16 + l16) * HID;
      v8bf blo = *(const v8bf*)(wrow + k0);
      v8bf bhi = *(const v8bf*)(wrow + k1);
      v16bf b = __builtin_shufflevector(blo, bhi, 0, 1, 2, 3, 4, 5, 6, 7,
                                        8, 9, 10, 11, 12, 13, 14, 15);
      acc[j] = __builtin_amdgcn_wmma_f32_16x16x32_bf16(
          false, a, false, b, (short)0, acc[j], false, false);
    }
  }

  // Epilogue: C/D layout — lane group selects M offset (0 or 8), VGPR r = M%8.
#pragma unroll
  for (int j = 0; j < 8; ++j) {
    int col = j * 16 + l16;
    float bc = bias ? bias[col] : 0.f;
#pragma unroll
    for (int r = 0; r < 8; ++r) {
      int m = r + (halfSel ? 8 : 0);
      size_t o = (size_t)(row0 + m) * HID + col;
      float val = acc[j][r] + bc;
      if (doRelu) val = val > 0.f ? val : 0.f;
      if (resid) val += resid[o];
      outf[o] = val;
      if (outh) outh[o] = (__bf16)val;
    }
  }
}

// ---------------------------------------------------------------------------
// Attention scores: one thread per (edge, head). 8 consecutive threads cover
// one edge's full 512B q/k rows -> coalesced gathers. exp(leaky_relu(q.k)),
// native f32 atomic add into denom.
// ---------------------------------------------------------------------------
__global__ __launch_bounds__(256) void att_score_kernel(
    const float* __restrict__ q, const float* __restrict__ k,
    const int* __restrict__ idx_kj, const int* __restrict__ idx_ji,
    float* __restrict__ att, float* __restrict__ denom) {
  int t = blockIdx.x * blockDim.x + threadIdx.x;
  if (t >= MEDGES * HEADS) return;
  int m = t >> 3, h = t & 7;
  int kj = idx_kj[m], ji = idx_ji[m];
  const float4* qp = (const float4*)(q + (size_t)kj * HID + h * DH);
  const float4* kp = (const float4*)(k + (size_t)ji * HID + h * DH);
  float s = 0.f;
#pragma unroll
  for (int i = 0; i < 4; ++i) {
    float4 a = qp[i], b = kp[i];
    s += a.x * b.x + a.y * b.y + a.z * b.z + a.w * b.w;
  }
  s = s > 0.f ? s : 0.2f * s;      // leaky_relu, slope 0.2
  float e = __expf(s);             // unsafe softmax numerator, as in reference
  att[t] = e;
  __hip_atomic_fetch_add(denom + (size_t)ji * HEADS + h, e,
                         __ATOMIC_RELAXED, __HIP_MEMORY_SCOPE_AGENT);
}

// ---------------------------------------------------------------------------
// Normalize + weighted-V scatter: one thread per (edge, head); 16 native f32
// atomic adds into agg[ji] per thread.
// ---------------------------------------------------------------------------
__global__ __launch_bounds__(256) void scatter_v_kernel(
    const float* __restrict__ v, const float* __restrict__ att,
    const float* __restrict__ denom,
    const int* __restrict__ idx_kj, const int* __restrict__ idx_ji,
    float* __restrict__ agg) {
  int t = blockIdx.x * blockDim.x + threadIdx.x;
  if (t >= MEDGES * HEADS) return;
  int m = t >> 3, h = t & 7;
  int kj = idx_kj[m], ji = idx_ji[m];
  float a = att[t] / denom[(size_t)ji * HEADS + h];
  const float4* vp = (const float4*)(v + (size_t)kj * HID + h * DH);
  float* ap = agg + (size_t)ji * HID + h * DH;
#pragma unroll
  for (int i = 0; i < 4; ++i) {
    float4 x = vp[i];
    __hip_atomic_fetch_add(ap + 4 * i + 0, x.x * a, __ATOMIC_RELAXED, __HIP_MEMORY_SCOPE_AGENT);
    __hip_atomic_fetch_add(ap + 4 * i + 1, x.y * a, __ATOMIC_RELAXED, __HIP_MEMORY_SCOPE_AGENT);
    __hip_atomic_fetch_add(ap + 4 * i + 2, x.z * a, __ATOMIC_RELAXED, __HIP_MEMORY_SCOPE_AGENT);
    __hip_atomic_fetch_add(ap + 4 * i + 3, x.w * a, __ATOMIC_RELAXED, __HIP_MEMORY_SCOPE_AGENT);
  }
}

// ---------------------------------------------------------------------------
extern "C" void kernel_launch(void* const* d_in, const int* in_sizes, int n_in,
                              void* d_out, int out_size, void* d_ws, size_t ws_size,
                              hipStream_t stream) {
  (void)in_sizes; (void)n_in; (void)out_size; (void)ws_size;

  const float* feats  = (const float*)d_in[0];
  const int*   idx_kj = (const int*)d_in[1];
  const int*   idx_ji = (const int*)d_in[2];
  const float* Wv = (const float*)d_in[3];
  const float* Wq = (const float*)d_in[4];
  const float* Wk = (const float*)d_in[5];
  const float* W1 = (const float*)d_in[6];
  const float* b1 = (const float*)d_in[7];
  const float* W2 = (const float*)d_in[8];
  const float* b2 = (const float*)d_in[9];

  // --- workspace carve-up (256B aligned) ---
  char* ws = (char*)d_ws;
  size_t off = 0;
  auto carve = [&](size_t bytes) -> char* {
    char* p = ws + off;
    off += (bytes + 255) & ~(size_t)255;
    return p;
  };
  const size_t SZ_ACT_BF = (size_t)NROWS * HID * sizeof(__bf16);   // 25.6 MB
  const size_t SZ_ACT_F  = (size_t)NROWS * HID * sizeof(float);    // 51.2 MB
  const size_t SZ_W_BF   = (size_t)HID * HID * sizeof(__bf16);     // 32 KB
  const size_t SZ_ATT    = (size_t)MEDGES * HEADS * sizeof(float); // 25.6 MB
  const size_t SZ_DEN    = (size_t)NROWS * HEADS * sizeof(float);  // 3.2 MB

  __bf16* featsh = (__bf16*)carve(SZ_ACT_BF);
  __bf16* Wvh    = (__bf16*)carve(SZ_W_BF);
  __bf16* Wqh    = (__bf16*)carve(SZ_W_BF);
  __bf16* Wkh    = (__bf16*)carve(SZ_W_BF);
  __bf16* W1h    = (__bf16*)carve(SZ_W_BF);
  __bf16* W2h    = (__bf16*)carve(SZ_W_BF);
  float*  qbuf   = (float*)carve(SZ_ACT_F);
  float*  kbuf   = (float*)carve(SZ_ACT_F);
  float*  vbuf   = (float*)carve(SZ_ACT_F);
  float*  attb   = (float*)carve(SZ_ATT);
  float*  denb   = (float*)carve(SZ_DEN);
  float*  aggb   = (float*)carve(SZ_ACT_F);
  // Aliases for dead buffers (producers finished before consumers start):
  float*  h1f    = qbuf;                            // f32 hidden of MLP layer 1
  __bf16* aggh   = (__bf16*)kbuf;                   // bf16 agg (first half of kbuf)
  __bf16* h1h    = (__bf16*)((char*)kbuf + SZ_ACT_BF); // bf16 hidden (second half)

  const int actN = NROWS * HID;  // 12.8M
  const int wN   = HID * HID;    // 16384
  const int edgeT = MEDGES * HEADS;

  // 1) convert feats + weights to bf16
  f32_to_bf16_kernel<<<(actN + 255) / 256, 256, 0, stream>>>(feats, featsh, actN);
  f32_to_bf16_kernel<<<(wN + 255) / 256, 256, 0, stream>>>(Wv, Wvh, wN);
  f32_to_bf16_kernel<<<(wN + 255) / 256, 256, 0, stream>>>(Wq, Wqh, wN);
  f32_to_bf16_kernel<<<(wN + 255) / 256, 256, 0, stream>>>(Wk, Wkh, wN);
  f32_to_bf16_kernel<<<(wN + 255) / 256, 256, 0, stream>>>(W1, W1h, wN);
  f32_to_bf16_kernel<<<(wN + 255) / 256, 256, 0, stream>>>(W2, W2h, wN);

  // 2) zero accumulation targets
  hipMemsetAsync(denb, 0, SZ_DEN, stream);
  hipMemsetAsync(aggb, 0, SZ_ACT_F, stream);

  // 3) q/k/v projections (WMMA bf16 -> f32)
  const int gemmBlocks = (NTILES + 3) / 4;  // 4 waves (128 threads) per block
  gemm_bf16_128<<<gemmBlocks, 128, 0, stream>>>(featsh, Wqh, nullptr, nullptr,
                                                qbuf, nullptr, NTILES, 0);
  gemm_bf16_128<<<gemmBlocks, 128, 0, stream>>>(featsh, Wkh, nullptr, nullptr,
                                                kbuf, nullptr, NTILES, 0);
  gemm_bf16_128<<<gemmBlocks, 128, 0, stream>>>(featsh, Wvh, nullptr, nullptr,
                                                vbuf, nullptr, NTILES, 0);

  // 4) edge attention scores + denom
  att_score_kernel<<<(edgeT + 255) / 256, 256, 0, stream>>>(
      qbuf, kbuf, idx_kj, idx_ji, attb, denb);

  // 5) normalize + weighted-V scatter into agg
  scatter_v_kernel<<<(edgeT + 255) / 256, 256, 0, stream>>>(
      vbuf, attb, denb, idx_kj, idx_ji, aggb);

  // 6) agg -> bf16 (q/k buffers are dead; aggh aliases kbuf)
  f32_to_bf16_kernel<<<(actN + 255) / 256, 256, 0, stream>>>(aggb, aggh, actN);

  // 7) MLP layer 1: relu(agg @ W1^T + b1), dual f32/bf16 output
  gemm_bf16_128<<<gemmBlocks, 128, 0, stream>>>(aggh, W1h, b1, nullptr,
                                                h1f, h1h, NTILES, 1);

  // 8) MLP layer 2 + residual: v + relu(h1 @ W2^T + b2) -> d_out (f32)
  gemm_bf16_128<<<gemmBlocks, 128, 0, stream>>>(h1h, W2h, b2, vbuf,
                                                (float*)d_out, nullptr, NTILES, 1);
}
